// MultiHeadAttention_33603824124653
// MI455X (gfx1250) — matrix operand order repaired and checked by reading, not compiled, not run on hardware
//
#include <hip/hip_runtime.h>

// ---------------------------------------------------------------------------
// MHA (B=2, S=2048, D=2048, H=16, Dh=128) for gfx1250.
// bf16 WMMA (v_wmma_f32_16x16x32_bf16) everywhere; f32 accumulation.
// Pipeline: f32->bf16 convert -> Q/K/V GEMMs (V stored transposed) -> RoPE
//           -> flash attention (online softmax, per-wave 16-row tiles)
//           -> O projection GEMM + bias (f32 out).
// ---------------------------------------------------------------------------

typedef __attribute__((ext_vector_type(16))) __bf16 v16bf;
typedef __attribute__((ext_vector_type(8)))  float  v8f;

#define HID   2048
#define SEQ   2048
#define NBATCH 2
#define NHEADS 16
#define HDIM  128
#define MTOT  4096   // B*S

__device__ __forceinline__ __bf16 f2bf(float x) {
  union { float f; unsigned u; } a; a.f = x;
  unsigned r = (a.u + 0x7fffu + ((a.u >> 16) & 1u)) >> 16;
  unsigned short h = (unsigned short)r;
  __bf16 o; __builtin_memcpy(&o, &h, 2); return o;
}
__device__ __forceinline__ float bf2f(__bf16 x) {
  unsigned short h; __builtin_memcpy(&h, &x, 2);
  unsigned u = ((unsigned)h) << 16;
  float f; __builtin_memcpy(&f, &u, 4); return f;
}

// A-matrix fragment (16x32 bf16): one contiguous 16-half run per lane
// at p (p already includes the hi*16 k-offset).
__device__ __forceinline__ v16bf load_a_frag(const __bf16* p) {
  union { uint4 u[2]; v16bf v; } t;
  t.u[0] = *(const uint4*)(p);
  t.u[1] = *(const uint4*)(p + 8);
  return t.v;
}
// B-matrix fragment (32x16 bf16): one contiguous 16-half run per lane
// (p already includes the hi*16 k-offset and the column selection).
__device__ __forceinline__ v16bf load_b_frag(const __bf16* p) {
  union { uint4 u[2]; v16bf v; } t;
  t.u[0] = *(const uint4*)(p);
  t.u[1] = *(const uint4*)(p + 8);
  return t.v;
}

__device__ __forceinline__ v8f wmma_bf16(v16bf a, v16bf b, v8f c) {
  return __builtin_amdgcn_wmma_f32_16x16x32_bf16(false, a, false, b, (short)0, c,
                                                 false, false);
}

// ---------------------------------------------------------------------------
__global__ void cvt_f32_bf16(const float* __restrict__ src,
                             __bf16* __restrict__ dst, int n) {
  int i = blockIdx.x * blockDim.x + threadIdx.x;
  if (i < n) dst[i] = f2bf(src[i]);
}

// ---------------------------------------------------------------------------
// C[m,n] = sum_k A[m,k] * W[n,k]   (A: [4096,2048] bf16, W: [2048,2048] bf16)
// mode 0: outb[m*HID+n] = bf16(C)
// mode 1: outb[(b*HID + n)*SEQ + s] = bf16(C)   (V transposed layout, b=m>>11)
// mode 2: outf[m*HID+n] = C + bias[n]           (f32 final output)
// Each wave: 32(M) x 64(N) tile. 8 waves / block.
__global__ __launch_bounds__(256) void gemm_nt(const __bf16* __restrict__ A,
                                               const __bf16* __restrict__ W,
                                               __bf16* __restrict__ outb,
                                               float* __restrict__ outf,
                                               const float* __restrict__ bias,
                                               int mode) {
  const int wave = threadIdx.x >> 5;
  const int lane = threadIdx.x & 31;
  const int mr = lane & 15;
  const int hi = lane >> 4;
  const int gw = blockIdx.x * 8 + wave;        // [0, 4096)
  const int NT = HID / 64;                     // 32
  const int mt = gw / NT;
  const int nt = gw % NT;
  const int row0 = mt * 32;
  const int col0 = nt * 64;

  v8f vzero = {};
  v8f acc[2][4];
#pragma unroll
  for (int i = 0; i < 2; ++i)
#pragma unroll
    for (int j = 0; j < 4; ++j) acc[i][j] = vzero;

  const __bf16* arow0 = A + (size_t)(row0 + mr) * HID + hi * 16;
  const __bf16* arow1 = arow0 + (size_t)16 * HID;
  const __bf16* wrow  = W + (size_t)(col0 + mr) * HID + hi * 16;

  for (int kk = 0; kk < HID; kk += 32) {
    v16bf a0 = load_a_frag(arow0 + kk);
    v16bf a1 = load_a_frag(arow1 + kk);
#pragma unroll
    for (int nb = 0; nb < 4; ++nb) {
      v16bf bf = load_b_frag(wrow + (size_t)(nb * 16) * HID + kk);
      acc[0][nb] = wmma_bf16(a0, bf, acc[0][nb]);
      acc[1][nb] = wmma_bf16(a1, bf, acc[1][nb]);
    }
  }

#pragma unroll
  for (int mi = 0; mi < 2; ++mi) {
#pragma unroll
    for (int nb = 0; nb < 4; ++nb) {
#pragma unroll
      for (int r = 0; r < 8; ++r) {
        int m = row0 + mi * 16 + r + 8 * hi;   // C layout: elem r -> row r+8*hi
        int c = col0 + nb * 16 + mr;           //           lane  -> col mr
        float v = acc[mi][nb][r];
        if (mode == 0) {
          outb[(size_t)m * HID + c] = f2bf(v);
        } else if (mode == 1) {
          outb[((size_t)((m >> 11) * HID + c)) * SEQ + (size_t)(m & (SEQ - 1))] =
              f2bf(v);
        } else {
          outf[(size_t)m * HID + c] = v + bias[c];
        }
      }
    }
  }
}

// ---------------------------------------------------------------------------
// In-place RoPE on Q and K (bf16, layout [m=b*S+s, h*128+d]).
// One thread per (m, h, d<64) handles the (d, d+64) rotation pair.
__global__ void rope_kernel(__bf16* __restrict__ Q, __bf16* __restrict__ K) {
  int t = blockIdx.x * blockDim.x + threadIdx.x;  // [0, 4194304)
  int m = t >> 10;
  int r = t & 1023;
  int h = r >> 6;
  int d = r & 63;
  int s = m & (SEQ - 1);
  size_t base = (size_t)m * HID + h * HDIM;
  // inv_freq = 10000^(-d/64) = exp(-d * ln(10000)/64)
  float ang = (float)s * __expf((float)d * -0.14391157f);
  float c, sn;
  __sincosf(ang, &sn, &c);
  float q1 = bf2f(Q[base + d]);
  float q2 = bf2f(Q[base + d + 64]);
  Q[base + d]      = f2bf(q1 * c - q2 * sn);
  Q[base + d + 64] = f2bf(q2 * c + q1 * sn);
  float k1 = bf2f(K[base + d]);
  float k2 = bf2f(K[base + d + 64]);
  K[base + d]      = f2bf(k1 * c - k2 * sn);
  K[base + d + 64] = f2bf(k2 * c + k1 * sn);
}

// ---------------------------------------------------------------------------
// Flash attention. One wave = one 16-query tile of one (b,h).
// Q,K: [m, h*128+d] bf16.  Vt: [(b*2048 + h*128 + d), s] bf16.
// Key blocks of 32; online softmax; P transposed via per-wave LDS (1KB).
__global__ __launch_bounds__(256) void flash_attn(const __bf16* __restrict__ Q,
                                                  const __bf16* __restrict__ K,
                                                  const __bf16* __restrict__ Vt,
                                                  __bf16* __restrict__ O) {
  __shared__ __bf16 pbuf[8 * 16 * 32];
  const int wave = threadIdx.x >> 5;
  const int lane = threadIdx.x & 31;
  const int mr = lane & 15;
  const int hi = lane >> 4;
  const int gw = blockIdx.x * 8 + wave;   // [0, 4096) = B*H*(S/16)
  const int mt = gw & 127;
  const int h  = (gw >> 7) & 15;
  const int b  = gw >> 11;
  const int row0 = b * SEQ + mt * 16;

  // Q fragments: 16x128 as 4 A-frags of 16x32
  v16bf qf[4];
  const __bf16* qp = Q + (size_t)(row0 + mr) * HID + h * HDIM + hi * 16;
#pragma unroll
  for (int c = 0; c < 4; ++c) qf[c] = load_a_frag(qp + c * 32);

  v8f vzero = {};
  v8f acc[8];
#pragma unroll
  for (int nb = 0; nb < 8; ++nb) acc[nb] = vzero;
  float rmax[8], rsum[8];
#pragma unroll
  for (int r = 0; r < 8; ++r) { rmax[r] = -3.0e38f; rsum[r] = 0.0f; }

  const float scale = 0.08838834764831845f;  // 1/sqrt(128)
  const __bf16* kb = K + (size_t)(b * SEQ) * HID + h * HDIM + hi * 16;
  const __bf16* vb = Vt + ((size_t)(b * SEQ + h * HDIM)) * SEQ + hi * 16;
  __bf16* pw = pbuf + wave * 512;                 // this wave's 16x32 P tile
  const __bf16* prd = pw + mr * 32 + hi * 16;     // A-frag read base

  for (int j = 0; j < SEQ / 32; ++j) {
    const int key0 = j * 32;
    // scores: S0 = keys key0..key0+15, S1 = keys key0+16..key0+31
    v8f s0 = vzero, s1 = vzero;
#pragma unroll
    for (int c = 0; c < 4; ++c) {
      v16bf b0 = load_b_frag(kb + (size_t)(key0 + mr) * HID + c * 32);
      v16bf b1 = load_b_frag(kb + (size_t)(key0 + 16 + mr) * HID + c * 32);
      s0 = wmma_bf16(qf[c], b0, s0);
      s1 = wmma_bf16(qf[c], b1, s1);
    }
    // online softmax; C-layout: element r -> row (r+8*hi), lane -> column
#pragma unroll
    for (int r = 0; r < 8; ++r) {
      float a0 = s0[r] * scale;
      float a1 = s1[r] * scale;
      float t = fmaxf(a0, a1);
      t = fmaxf(t, __shfl_xor(t, 1));
      t = fmaxf(t, __shfl_xor(t, 2));
      t = fmaxf(t, __shfl_xor(t, 4));
      t = fmaxf(t, __shfl_xor(t, 8));
      float nm = fmaxf(rmax[r], t);
      float corr = __expf(rmax[r] - nm);
      rmax[r] = nm;
      float p0 = __expf(a0 - nm);
      float p1 = __expf(a1 - nm);
      float ps = p0 + p1;
      ps += __shfl_xor(ps, 1);
      ps += __shfl_xor(ps, 2);
      ps += __shfl_xor(ps, 4);
      ps += __shfl_xor(ps, 8);
      rsum[r] = rsum[r] * corr + ps;
#pragma unroll
      for (int nb = 0; nb < 8; ++nb) acc[nb][r] *= corr;
      // stage P (C-layout -> LDS row-major 16x32)
      pw[(r + 8 * hi) * 32 + mr]      = f2bf(p0);
      pw[(r + 8 * hi) * 32 + mr + 16] = f2bf(p1);
    }
    asm volatile("" ::: "memory");   // keep DS stores before DS loads
    v16bf pa = load_a_frag(prd);     // P as A-frag (16x32); DS in-order per wave
    asm volatile("" ::: "memory");
    // out += P x V  (8 WMMAs over the 128-wide head dim)
#pragma unroll
    for (int nb = 0; nb < 8; ++nb) {
      v16bf vf = load_b_frag(vb + (size_t)(nb * 16 + mr) * SEQ + key0);
      acc[nb] = wmma_bf16(pa, vf, acc[nb]);
    }
  }

  // finalize: divide by row sums, store [m, h*128+d] bf16
#pragma unroll
  for (int r = 0; r < 8; ++r) {
    float inv = 1.0f / rsum[r];
#pragma unroll
    for (int nb = 0; nb < 8; ++nb) {
      float v = acc[nb][r] * inv;
      O[(size_t)(row0 + r + 8 * hi) * HID + h * HDIM + nb * 16 + mr] = f2bf(v);
    }
  }
}

// ---------------------------------------------------------------------------
extern "C" void kernel_launch(void* const* d_in, const int* in_sizes, int n_in,
                              void* d_out, int out_size, void* d_ws,
                              size_t ws_size, hipStream_t stream) {
  const float* x  = (const float*)d_in[0];
  const float* Wq = (const float*)d_in[1];
  const float* Wk = (const float*)d_in[2];
  const float* Wv = (const float*)d_in[3];
  const float* Wo = (const float*)d_in[4];
  const float* bo = (const float*)d_in[5];
  float* out = (float*)d_out;

  const size_t XS = (size_t)MTOT * HID;   // 8,388,608 elements
  const size_t WSZ = (size_t)HID * HID;   // 4,194,304 elements

  char* p = (char*)d_ws;                  // 112 MB total
  __bf16* xb  = (__bf16*)p; p += XS * 2;
  __bf16* wqb = (__bf16*)p; p += WSZ * 2;
  __bf16* wkb = (__bf16*)p; p += WSZ * 2;
  __bf16* wvb = (__bf16*)p; p += WSZ * 2;
  __bf16* wob = (__bf16*)p; p += WSZ * 2;
  __bf16* qb  = (__bf16*)p; p += XS * 2;
  __bf16* kbm = (__bf16*)p; p += XS * 2;
  __bf16* vt  = (__bf16*)p; p += XS * 2;
  __bf16* ab  = (__bf16*)p; p += XS * 2;

  cvt_f32_bf16<<<(int)(XS / 256), 256, 0, stream>>>(x, xb, (int)XS);
  cvt_f32_bf16<<<(int)(WSZ / 256), 256, 0, stream>>>(Wq, wqb, (int)WSZ);
  cvt_f32_bf16<<<(int)(WSZ / 256), 256, 0, stream>>>(Wk, wkb, (int)WSZ);
  cvt_f32_bf16<<<(int)(WSZ / 256), 256, 0, stream>>>(Wv, wvb, (int)WSZ);
  cvt_f32_bf16<<<(int)(WSZ / 256), 256, 0, stream>>>(Wo, wob, (int)WSZ);

  const int gemm_blocks = (MTOT / 32) * (HID / 64) / 8;  // 512
  gemm_nt<<<gemm_blocks, 256, 0, stream>>>(xb, wqb, qb, nullptr, nullptr, 0);
  gemm_nt<<<gemm_blocks, 256, 0, stream>>>(xb, wkb, kbm, nullptr, nullptr, 0);
  gemm_nt<<<gemm_blocks, 256, 0, stream>>>(xb, wvb, vt, nullptr, nullptr, 1);

  rope_kernel<<<(MTOT * NHEADS * 64) / 256, 256, 0, stream>>>(qb, kbm);

  flash_attn<<<(NBATCH * NHEADS * (SEQ / 16)) / 8, 256, 0, stream>>>(qb, kbm,
                                                                     vt, ab);

  gemm_nt<<<gemm_blocks, 256, 0, stream>>>(ab, wob, nullptr, out, bo, 2);
}
